// SwinTransformerBlock_61838939127890
// MI455X (gfx1250) — matrix operand order repaired and checked by reading, not compiled, not run on hardware
//
#include <hip/hip_runtime.h>
#include <hip/hip_bf16.h>

typedef _Float16 v16h __attribute__((ext_vector_type(16)));
typedef _Float16 v8h  __attribute__((ext_vector_type(8)));
typedef _Float16 v4h  __attribute__((ext_vector_type(4)));
typedef float    v8f  __attribute__((ext_vector_type(8)));

#define WMMA_F32_F16(a, b, c) \
  __builtin_amdgcn_wmma_f32_16x16x32_f16(false, (a), false, (b), (short)0, (c), false, false)

__device__ __forceinline__ v8f zero8() {
  v8f z = {0.f, 0.f, 0.f, 0.f, 0.f, 0.f, 0.f, 0.f};
  return z;
}

// A-style fragment (16x32 f16, per ISA layout): 8 halves at p, 8 halves at p+16.
__device__ __forceinline__ v16h load_frag_8_8(const _Float16* p) {
  v8h lo = *(const v8h*)p;
  v8h hi = *(const v8h*)(p + 16);
  return __builtin_shufflevector(lo, hi, 0, 1, 2, 3, 4, 5, 6, 7, 8, 9, 10, 11, 12, 13, 14, 15);
}
// B-style fragment (32x16 f16): 8 halves at p, 8 halves at p+16.
__device__ __forceinline__ v16h load_frag_16(const _Float16* p) {
  v8h lo = *(const v8h*)p;
  v8h hi = *(const v8h*)(p + 16);
  return __builtin_shufflevector(lo, hi, 0, 1, 2, 3, 4, 5, 6, 7, 8, 9, 10, 11, 12, 13, 14, 15);
}

__device__ __forceinline__ float wave_sum(float v) {
#pragma unroll
  for (int m = 16; m > 0; m >>= 1) v += __shfl_xor(v, m, 32);
  return v;
}

// Async global->LDS 16-byte copy (CDNA5 ASYNCcnt path). The low 32 bits of a
// generic pointer into LDS are the wave-relative LDS byte offset (ISA 10.2:
// flat LDS addressing truncates to addr[31:0]).
__device__ __forceinline__ void async_copy_b128(void* lds_dst, const void* gsrc) {
  uint32_t l = (uint32_t)(uintptr_t)lds_dst;
  asm volatile("global_load_async_to_lds_b128 %0, %1, off" ::"v"(l), "v"(gsrc) : "memory");
}

// ---------------------------------------------------------------------------
// Generic WMMA GEMM: C[M,N] = epi(A[M,K] @ W[N,K]^T + bias[N])
// MODE 0: f16 out; MODE 1: f32 out; MODE 2: f16 out with exact GELU
// Workgroup tile 128x128, 8 waves (2 M x 4 N), wave tile 64x32.
// A tile (128x32) staged through LDS with double-buffered async copies;
// B fragments loaded direct from global (+ prefetch of next K-step).
// ---------------------------------------------------------------------------
template <int MODE>
__global__ __launch_bounds__(256) void gemm_wmma_kernel(
    const _Float16* __restrict__ A, const _Float16* __restrict__ W,
    const float* __restrict__ bias, void* __restrict__ Cout, int N, int K) {
  __shared__ _Float16 ldsA[2][128 * 40];  // 2 x 10240 B, row stride 40 halves

  const int tid = threadIdx.x;
  const int lane = tid & 31;
  const int wave = tid >> 5;
  const int l16 = lane & 15;
  const int hi = lane >> 4;
  const long mblk = (long)blockIdx.y * 128;
  const int mwoff = (wave >> 2) * 64;
  const int nbase = blockIdx.x * 128 + (wave & 3) * 32;

  // issue async copies for one 128x32 A tile (512 x 16B chunks, 2 per thread)
  auto issue_tile = [&](int buf, int kb) {
#pragma unroll
    for (int q = 0; q < 2; q++) {
      int idx = tid + q * 256;
      int row = idx >> 2, c = idx & 3;
      async_copy_b128(&ldsA[buf][row * 40 + c * 8],
                      A + (mblk + row) * (long)K + kb + c * 8);
    }
  };

  v8f acc[4][2];
#pragma unroll
  for (int i = 0; i < 4; i++) {
    acc[i][0] = zero8();
    acc[i][1] = zero8();
  }

  issue_tile(0, 0);
  int cur = 0;
  for (int kb = 0; kb < K; kb += 32) {
    const bool more = (kb + 32) < K;
    if (more) {
      issue_tile(cur ^ 1, kb + 32);
      // current tile's 2 async ops complete in-order ahead of the 2 just issued
      asm volatile("s_wait_asynccnt 0x2" ::: "memory");
    } else {
      asm volatile("s_wait_asynccnt 0x0" ::: "memory");
    }
    __syncthreads();  // tile visible to all waves

    v16h bf[2];
#pragma unroll
    for (int j = 0; j < 2; j++) {
      const _Float16* wp = W + (long)(nbase + j * 16 + l16) * K + kb + hi * 8;
      bf[j] = load_frag_16(wp);
      if (more) __builtin_prefetch(wp + 32, 0, 1);  // next K-step B rows
    }
    v16h af[4];
#pragma unroll
    for (int i = 0; i < 4; i++)
      af[i] = load_frag_8_8(&ldsA[cur][(mwoff + i * 16 + l16) * 40 + hi * 8]);
#pragma unroll
    for (int i = 0; i < 4; i++)
#pragma unroll
      for (int j = 0; j < 2; j++) acc[i][j] = WMMA_F32_F16(af[i], bf[j], acc[i][j]);

    __syncthreads();  // all waves done reading before buffer is overwritten
    cur ^= 1;
  }

#pragma unroll
  for (int i = 0; i < 4; i++)
#pragma unroll
    for (int j = 0; j < 2; j++)
#pragma unroll
      for (int r = 0; r < 8; r++) {
        long row = mblk + mwoff + i * 16 + r + hi * 8;
        int col = nbase + j * 16 + l16;
        float v = acc[i][j][r] + bias[col];
        if (MODE == 2) v = 0.5f * v * (1.0f + erff(v * 0.7071067811865475f));
        if (MODE == 1)
          ((float*)Cout)[row * (long)N + col] = v;
        else
          ((_Float16*)Cout)[row * (long)N + col] = (_Float16)v;
      }
}

// ---------------------------------------------------------------------------
// Windowed cosine attention: one wave per (window, head). 4 waves per block,
// each wave iterates 4 heads. Per-wave private LDS region (32256 B):
//   qn[64][40]h, kn[64][40]h, vT[32][72]h, sS[64][68]f ; P aliases qn.
// ---------------------------------------------------------------------------
__global__ __launch_bounds__(128) void attn_kernel(
    const _Float16* __restrict__ qkv,       // [131072][1536]
    const float* __restrict__ logit_scale,  // [16]
    const float* __restrict__ biastab,      // [16][64][64]
    _Float16* __restrict__ outh) {          // [131072][512]
  extern __shared__ char smem[];
  const int lane = threadIdx.x & 31;
  const int wave = threadIdx.x >> 5;
  const int l16 = lane & 15;
  const int hi = lane >> 4;
  const int g = blockIdx.x;  // global window 0..2047
  const int wy = (g & 63) >> 3, wx = g & 7;
  const long rowbase = (long)g << 6;

  char* base = smem + wave * 32256;
  _Float16* qn = (_Float16*)base;      // 64*40 halves
  _Float16* kn = qn + 64 * 40;         // 64*40 halves
  _Float16* vT = kn + 64 * 40;         // 32*72 halves
  float* sS = (float*)(vT + 32 * 72);  // 64*68 floats
  _Float16* pP = qn;                   // [64][72], aliases qn (+into kn)

  for (int h = wave; h < 16; h += 4) {
    const float scale = __expf(fminf(logit_scale[h], 4.6051701859880914f));

    // --- normalize q,k rows; transpose v into vT ---
#pragma unroll
    for (int rr = 0; rr < 2; rr++) {
      int r = lane + rr * 32;
      const _Float16* qp = qkv + (rowbase + r) * 1536 + h * 32;
      v8h q8[4], k8[4], vv[4];
#pragma unroll
      for (int c = 0; c < 4; c++) {
        q8[c] = *(const v8h*)(qp + c * 8);
        k8[c] = *(const v8h*)(qp + 512 + c * 8);
        vv[c] = *(const v8h*)(qp + 1024 + c * 8);
      }
      float qs = 0.f, ks = 0.f;
#pragma unroll
      for (int c = 0; c < 4; c++)
#pragma unroll
        for (int d = 0; d < 8; d++) {
          float q = (float)q8[c][d], k = (float)k8[c][d];
          qs += q * q;
          ks += k * k;
        }
      float qi = 1.0f / fmaxf(sqrtf(qs), 1e-12f);
      float ki = 1.0f / fmaxf(sqrtf(ks), 1e-12f);
#pragma unroll
      for (int c = 0; c < 4; c++)
#pragma unroll
        for (int d = 0; d < 8; d++) {
          qn[r * 40 + c * 8 + d] = (_Float16)((float)q8[c][d] * qi);
          kn[r * 40 + c * 8 + d] = (_Float16)((float)k8[c][d] * ki);
          vT[(c * 8 + d) * 72 + r] = vv[c][d];
        }
    }

    // --- S = qn @ kn^T   (64x64, K=32: 16 WMMAs) ---
    v16h af[4];
#pragma unroll
    for (int i = 0; i < 4; i++) af[i] = load_frag_8_8(qn + (i * 16 + l16) * 40 + hi * 8);
#pragma unroll
    for (int jh = 0; jh < 2; jh++) {
      v16h bf[2];
#pragma unroll
      for (int j = 0; j < 2; j++)
        bf[j] = load_frag_16(kn + ((jh * 2 + j) * 16 + l16) * 40 + hi * 8);
#pragma unroll
      for (int i = 0; i < 4; i++)
#pragma unroll
        for (int j = 0; j < 2; j++) {
          v8f s = WMMA_F32_F16(af[i], bf[j], zero8());
#pragma unroll
          for (int r = 0; r < 8; r++)
            sS[(i * 16 + r + hi * 8) * 68 + (jh * 2 + j) * 16 + l16] = s[r];
        }
    }

    // --- scale + CPB bias + shift mask + softmax -> P (f16) ---
#pragma unroll
    for (int rr = 0; rr < 2; rr++) {
      int i = lane + rr * 32;
      int hrow = wy * 8 + (i >> 3), wcol = wx * 8 + (i & 7);
      int ri = (hrow < 56 ? 0 : (hrow < 60 ? 1 : 2)) * 3 +
               (wcol < 56 ? 0 : (wcol < 60 ? 1 : 2));
      const float* brow = biastab + ((h << 12) + (i << 6));
      float mx = -1e30f;
      for (int j = 0; j < 64; j++) {
        int hj = wy * 8 + (j >> 3), wj = wx * 8 + (j & 7);
        int rj = (hj < 56 ? 0 : (hj < 60 ? 1 : 2)) * 3 + (wj < 56 ? 0 : (wj < 60 ? 1 : 2));
        float v = sS[i * 68 + j] * scale + brow[j] + (ri == rj ? 0.f : -100.f);
        sS[i * 68 + j] = v;
        mx = fmaxf(mx, v);
      }
      float sum = 0.f;
      for (int j = 0; j < 64; j++) {
        float e = __expf(sS[i * 68 + j] - mx);
        sum += e;
        sS[i * 68 + j] = e;
      }
      float inv = 1.0f / sum;
      for (int j = 0; j < 64; j++) pP[i * 72 + j] = (_Float16)(sS[i * 68 + j] * inv);
    }

    // --- O = P @ V  (64x32, K=64: 16 WMMAs) ---
    v8f oacc[4][2];
#pragma unroll
    for (int i = 0; i < 4; i++) {
      oacc[i][0] = zero8();
      oacc[i][1] = zero8();
    }
#pragma unroll
    for (int ks = 0; ks < 2; ks++) {
      int kk = ks * 32;
      v16h pf[4], vf[2];
#pragma unroll
      for (int i = 0; i < 4; i++)
        pf[i] = load_frag_8_8(pP + (i * 16 + l16) * 72 + kk + hi * 8);
#pragma unroll
      for (int j = 0; j < 2; j++)
        vf[j] = load_frag_16(vT + (j * 16 + l16) * 72 + kk + hi * 8);
#pragma unroll
      for (int i = 0; i < 4; i++)
#pragma unroll
        for (int j = 0; j < 2; j++) oacc[i][j] = WMMA_F32_F16(pf[i], vf[j], oacc[i][j]);
    }
#pragma unroll
    for (int i = 0; i < 4; i++)
#pragma unroll
      for (int j = 0; j < 2; j++)
#pragma unroll
        for (int r = 0; r < 8; r++) {
          int tok = i * 16 + r + hi * 8;
          outh[(rowbase + tok) * 512 + h * 32 + j * 16 + l16] = (_Float16)oacc[i][j][r];
        }
  }
}

// ---------------------------------------------------------------------------
// Fused LayerNorm + residual (optionally gathering through window-reverse +
// un-roll). One wave per row of 512 channels.
// ---------------------------------------------------------------------------
__global__ __launch_bounds__(256) void norm_res_kernel(
    const float* __restrict__ y, const float* __restrict__ res,
    const float* __restrict__ w, const float* __restrict__ b,
    float* __restrict__ outf, _Float16* __restrict__ outh, int permute) {
  const int lane = threadIdx.x & 31;
  const long row = (long)blockIdx.x * 8 + (threadIdx.x >> 5);
  long srow = row;
  if (permute) {
    int bb = (int)(row >> 12), hw = (int)(row & 4095);
    int hh = hw >> 6, ww = hw & 63;
    int hp = (hh + 60) & 63, wp = (ww + 60) & 63;  // un-roll by +SHIFT
    srow = ((long)(bb * 64 + (hp >> 3) * 8 + (wp >> 3)) << 6) + (hp & 7) * 8 + (wp & 7);
  }
  const float* src = y + srow * 512;
  float vals[16];
  float s = 0.f, s2 = 0.f;
#pragma unroll
  for (int k = 0; k < 4; k++) {
    float4 v4 = *(const float4*)(src + (lane + 32 * k) * 4);
    vals[k * 4 + 0] = v4.x;
    vals[k * 4 + 1] = v4.y;
    vals[k * 4 + 2] = v4.z;
    vals[k * 4 + 3] = v4.w;
    s += v4.x + v4.y + v4.z + v4.w;
    s2 += v4.x * v4.x + v4.y * v4.y + v4.z * v4.z + v4.w * v4.w;
  }
  s = wave_sum(s);
  s2 = wave_sum(s2);
  float mu = s * (1.0f / 512.0f);
  float var = s2 * (1.0f / 512.0f) - mu * mu;
  float rstd = rsqrtf(var + 1e-5f);
#pragma unroll
  for (int k = 0; k < 4; k++) {
    int c = (lane + 32 * k) * 4;
    float4 rv = *(const float4*)(res + row * 512 + c);
    float o0 = rv.x + (vals[k * 4 + 0] - mu) * rstd * w[c + 0] + b[c + 0];
    float o1 = rv.y + (vals[k * 4 + 1] - mu) * rstd * w[c + 1] + b[c + 1];
    float o2 = rv.z + (vals[k * 4 + 2] - mu) * rstd * w[c + 2] + b[c + 2];
    float o3 = rv.w + (vals[k * 4 + 3] - mu) * rstd * w[c + 3] + b[c + 3];
    float4 of = {o0, o1, o2, o3};
    *(float4*)(outf + row * 512 + c) = of;
    if (outh != nullptr) {
      v4h oh = {(_Float16)o0, (_Float16)o1, (_Float16)o2, (_Float16)o3};
      *(v4h*)(outh + row * 512 + c) = oh;
    }
  }
}

// --- shifted-window gather + f32->f16: x -> xwin[131072][512] ---
__global__ __launch_bounds__(256) void xwin_kernel(const float* __restrict__ x,
                                                   _Float16* __restrict__ xw) {
  int idx = blockIdx.x * 256 + threadIdx.x;  // 16,777,216 total
  int row = idx >> 7;
  int c4 = idx & 127;
  int g = row >> 6, tok = row & 63;
  int b = g >> 6, wi = g & 63;
  int hh = (((wi >> 3) << 3) + (tok >> 3) + 4) & 63;  // roll by -SHIFT
  int ww = (((wi & 7) << 3) + (tok & 7) + 4) & 63;
  float4 v = *(const float4*)(x + ((long)b * 4096 + hh * 64 + ww) * 512 + c4 * 4);
  v4h o = {(_Float16)v.x, (_Float16)v.y, (_Float16)v.z, (_Float16)v.w};
  *(v4h*)(xw + (long)row * 512 + c4 * 4) = o;
}

// --- generic f32 -> f16 weight conversion (n multiple of 4) ---
__global__ __launch_bounds__(256) void cvt_kernel(const float* __restrict__ src,
                                                  _Float16* __restrict__ dst, int n4) {
  int i = blockIdx.x * 256 + threadIdx.x;
  if (i >= n4) return;
  float4 v = ((const float4*)src)[i];
  v4h o = {(_Float16)v.x, (_Float16)v.y, (_Float16)v.z, (_Float16)v.w};
  ((v4h*)dst)[i] = o;
}

// --- qkv bias assembly: [q_bias, 0, v_bias] ---
__global__ void qkvb_kernel(const float* __restrict__ qb, const float* __restrict__ vb,
                            float* __restrict__ out) {
  int i = blockIdx.x * 256 + threadIdx.x;
  if (i >= 1536) return;
  out[i] = i < 512 ? qb[i] : (i < 1024 ? 0.f : vb[i - 1024]);
}

// --- continuous relative position bias MLP: table[225][16] ---
__global__ void cpb_table_kernel(const float* __restrict__ w1, const float* __restrict__ b1,
                                 const float* __restrict__ w2, float* __restrict__ table) {
  __shared__ float red[256];
  int p = blockIdx.x;
  int t = threadIdx.x;
  int i = p / 15, j = p % 15;
  float r0 = (float)(i - 7) * (8.0f / 7.0f);
  float r1 = (float)(j - 7) * (8.0f / 7.0f);
  float t0 = (r0 >= 0.f ? 1.f : -1.f) * log2f(fabsf(r0) + 1.f) * (1.f / 3.f);
  float t1 = (r1 >= 0.f ? 1.f : -1.f) * log2f(fabsf(r1) + 1.f) * (1.f / 3.f);
  float ha = fmaxf(0.f, w1[t * 2] * t0 + w1[t * 2 + 1] * t1 + b1[t]);
  float hb = fmaxf(0.f, w1[(t + 256) * 2] * t0 + w1[(t + 256) * 2 + 1] * t1 + b1[t + 256]);
  for (int h = 0; h < 16; h++) {
    red[t] = ha * w2[h * 512 + t] + hb * w2[h * 512 + t + 256];
    __syncthreads();
    for (int s = 128; s > 0; s >>= 1) {
      if (t < s) red[t] += red[t + s];
      __syncthreads();
    }
    if (t == 0) table[p * 16 + h] = red[0];
    __syncthreads();
  }
}

// --- biastab[h][i][j] = 16*sigmoid(table[rpi(i,j)][h]) ---
__global__ void bias_build_kernel(const float* __restrict__ table, float* __restrict__ biastab) {
  int idx = blockIdx.x * 256 + threadIdx.x;  // 65536
  int h = idx >> 12;
  int rem = idx & 4095;
  int i = rem >> 6, j = rem & 63;
  int dy = (i >> 3) - (j >> 3) + 7;
  int dx = (i & 7) - (j & 7) + 7;
  float v = table[(dy * 15 + dx) * 16 + h];
  biastab[idx] = 16.0f / (1.0f + expf(-v));
}

// ---------------------------------------------------------------------------
extern "C" void kernel_launch(void* const* d_in, const int* in_sizes, int n_in,
                              void* d_out, int out_size, void* d_ws, size_t ws_size,
                              hipStream_t stream) {
  (void)in_sizes; (void)n_in; (void)out_size; (void)ws_size;
  const float* x = (const float*)d_in[0];
  const float* qkv_w = (const float*)d_in[1];
  const float* q_bias = (const float*)d_in[2];
  const float* v_bias = (const float*)d_in[3];
  const float* logit_scale = (const float*)d_in[4];
  const float* cpb_w1 = (const float*)d_in[5];
  const float* cpb_b1 = (const float*)d_in[6];
  const float* cpb_w2 = (const float*)d_in[7];
  const float* proj_w = (const float*)d_in[8];
  const float* proj_b = (const float*)d_in[9];
  const float* norm1_w = (const float*)d_in[10];
  const float* norm1_b = (const float*)d_in[11];
  const float* norm2_w = (const float*)d_in[12];
  const float* norm2_b = (const float*)d_in[13];
  const float* fc1_w = (const float*)d_in[14];
  const float* fc1_b = (const float*)d_in[15];
  const float* fc2_w = (const float*)d_in[16];
  const float* fc2_b = (const float*)d_in[17];

  const size_t M = 131072;  // B*NW*N tokens
  char* ws = (char*)d_ws;
  size_t off = 0;
  auto alloc = [&](size_t bytes) {
    size_t o = off;
    off += (bytes + 255) & ~(size_t)255;
    return o;
  };
  _Float16* r0 = (_Float16*)(ws + alloc(M * 512 * 2));     // xwin, later attn-out
  _Float16* qkvh = (_Float16*)(ws + alloc(M * 1536 * 2));  // qkv, later x1 f16
  float* gtmp = (float*)(ws + alloc(M * 512 * 4));         // proj-out, later fc2-out
  float* x1f = (float*)(ws + alloc(M * 512 * 4));
  _Float16* hbuf = (_Float16*)(ws + alloc(M * 2048 * 2));  // fc1 out f16
  _Float16* qkvwh = (_Float16*)(ws + alloc(1536 * 512 * 2));
  _Float16* projwh = (_Float16*)(ws + alloc(512 * 512 * 2));
  _Float16* fc1wh = (_Float16*)(ws + alloc(2048 * 512 * 2));
  _Float16* fc2wh = (_Float16*)(ws + alloc(512 * 2048 * 2));
  float* qkvb = (float*)(ws + alloc(1536 * 4));
  float* table = (float*)(ws + alloc(225 * 16 * 4));
  float* biastab = (float*)(ws + alloc(16 * 4096 * 4));
  _Float16* x1h = qkvh;    // alias: qkv dead after attention
  _Float16* attnout = r0;  // alias: xwin dead after qkv gemm

  // 0) weight / bias prep
  cvt_kernel<<<(1536 * 512 / 4 + 255) / 256, 256, 0, stream>>>(qkv_w, qkvwh, 1536 * 512 / 4);
  cvt_kernel<<<(512 * 512 / 4 + 255) / 256, 256, 0, stream>>>(proj_w, projwh, 512 * 512 / 4);
  cvt_kernel<<<(2048 * 512 / 4 + 255) / 256, 256, 0, stream>>>(fc1_w, fc1wh, 2048 * 512 / 4);
  cvt_kernel<<<(512 * 2048 / 4 + 255) / 256, 256, 0, stream>>>(fc2_w, fc2wh, 512 * 2048 / 4);
  qkvb_kernel<<<6, 256, 0, stream>>>(q_bias, v_bias, qkvb);
  cpb_table_kernel<<<225, 256, 0, stream>>>(cpb_w1, cpb_b1, cpb_w2, table);
  bias_build_kernel<<<256, 256, 0, stream>>>(table, biastab);

  // 1) roll + window partition + f16
  xwin_kernel<<<65536, 256, 0, stream>>>(x, r0);

  // 2) QKV GEMM: [131072,512] x [1536,512]^T -> f16
  gemm_wmma_kernel<0><<<dim3(12, 1024), 256, 0, stream>>>(r0, qkvwh, qkvb, (void*)qkvh, 1536, 512);

  // 3) windowed cosine attention (WMMA in LDS)
  attn_kernel<<<2048, 128, 4 * 32256, stream>>>(qkvh, logit_scale, biastab, attnout);

  // 4) proj GEMM -> f32 (window order)
  gemm_wmma_kernel<1><<<dim3(4, 1024), 256, 0, stream>>>(attnout, projwh, proj_b, (void*)gtmp, 512, 512);

  // 5) window-reverse + un-roll + LN + residual -> x1 (f32 + f16)
  norm_res_kernel<<<16384, 256, 0, stream>>>(gtmp, x, norm1_w, norm1_b, x1f, x1h, 1);

  // 6) fc1 GEMM + exact GELU -> f16
  gemm_wmma_kernel<2><<<dim3(16, 1024), 256, 0, stream>>>(x1h, fc1wh, fc1_b, (void*)hbuf, 2048, 512);

  // 7) fc2 GEMM -> f32
  gemm_wmma_kernel<1><<<dim3(4, 1024), 256, 0, stream>>>(hbuf, fc2wh, fc2_b, (void*)gtmp, 512, 2048);

  // 8) final LN + residual -> d_out (f32)
  norm_res_kernel<<<16384, 256, 0, stream>>>(gtmp, x1f, norm2_w, norm2_b, (float*)d_out, nullptr, 0);
}
